// BlockDiagonalLinear_text_5952824673029
// MI455X (gfx1250) — matrix operand order repaired and checked by reading, not compiled, not run on hardware
//
#include <hip/hip_runtime.h>
#include <hip/hip_bf16.h>
#include <math.h>

typedef __attribute__((ext_vector_type(16))) _Float16 v16h;
typedef __attribute__((ext_vector_type(8)))  _Float16 v8h;
typedef __attribute__((ext_vector_type(8)))  float    v8f;

#define R_BLK 16
#define BSZ   256
#define DDIM  4096
#define MTILE 16
#define SQRTC 0.1f

// ---------------------------------------------------------------------------
// Prep: convert block-diagonal weights f32 -> f16 (2 MB, stays L2-resident).
// ---------------------------------------------------------------------------
__global__ void wconv_kernel(const float* __restrict__ W,
                             _Float16* __restrict__ Wh, int n4) {
    int i = blockIdx.x * blockDim.x + threadIdx.x;
    if (i < n4) {
        float4 w = ((const float4*)W)[i];
        union { _Float16 h[4]; unsigned long long u; } p;
        p.h[0] = (_Float16)w.x; p.h[1] = (_Float16)w.y;
        p.h[2] = (_Float16)w.z; p.h[3] = (_Float16)w.w;
        ((unsigned long long*)Wh)[i] = p.u;
    }
}

// ---------------------------------------------------------------------------
// Fused expmap0 -> block-diag mobius matvec (WMMA) -> project -> logmap0.
// One workgroup (8 wave32) handles 16 full rows of 4096.
// ---------------------------------------------------------------------------
__global__ __launch_bounds__(256)
void hyp_blockdiag_kernel(const float* __restrict__ x,
                          const _Float16* __restrict__ Wh,
                          float* __restrict__ out)
{
    extern __shared__ __align__(16) char smem_raw[];
    float*    mx     = (float*)smem_raw;                         // [16][4096] f32
    float*    xss    = (float*)(smem_raw + MTILE * DDIM * 4);    // [16]
    float*    s_     = xss + 16;                                 // [16] expmap scale
    float*    alpha_ = s_ + 16;                                  // [16] artanh(c*xn)/xn
    float*    msum   = alpha_ + 16;                              // [16] sum mx^2
    float*    mscale = msum + 16;                                // [16] final scale
    unsigned* mnz    = (unsigned*)(mscale + 16);                 // [1] nonzero bitmask

    const int tid     = threadIdx.x;
    const int lane    = tid & 31;
    const int wave    = tid >> 5;
    const int rowBase = blockIdx.x * MTILE;

    if (tid < 16) msum[tid] = 0.f;
    if (tid == 0) mnz[0] = 0u;

    // ---- pass 1: per-row sum of squares of x (coalesced float4) ----
    {
        const int m = tid >> 4;          // row 0..15 (wave w: lanes 0-15 -> 2w, 16-31 -> 2w+1)
        const int c = tid & 15;
        const float4* xr = (const float4*)(x + (size_t)(rowBase + m) * DDIM);
        float acc = 0.f;
        #pragma unroll 4
        for (int j = 0; j < 64; ++j) {
            float4 v = xr[c + 16 * j];
            acc += v.x * v.x + v.y * v.y + v.z * v.z + v.w * v.w;
        }
        acc += __shfl_xor(acc, 1, 32);
        acc += __shfl_xor(acc, 2, 32);
        acc += __shfl_xor(acc, 4, 32);
        acc += __shfl_xor(acc, 8, 32);
        if ((lane & 15) == 0) xss[m] = acc;
    }
    __syncthreads();

    if (tid < 16) {
        float xn = sqrtf(xss[tid]);
        float un = fmaxf(xn, 1e-5f);
        float t  = tanhf(fminf(SQRTC * un, 15.f));
        float sc = t / (SQRTC * un);                 // h = sc * x
        s_[tid]  = sc;
        float xnorm = fmaxf(sc * xn, 1e-5f);         // ||h||
        float z0 = fminf(fmaxf(SQRTC * xnorm, -1.f + 1e-5f), 1.f - 1e-5f);
        float at = 0.5f * (logf(1.f + z0) - logf(1.f - z0));
        alpha_[tid] = at / xnorm;
    }
    __syncthreads();

    // ---- WMMA GEMM: each wave owns 2 diagonal blocks of 16 ----
    const int   nrow  = lane & 15;                 // A row / B col this lane holds
    const int   loffA = (lane & 16) ? 8 : 0;       // A: K in {loffA..+7, loffA+16..+23}
    const int   loffB = (lane & 16) ? 8 : 0;       // B: K in {loffB..+7, loffB+16..+23}
    const float sA    = s_[nrow];
    const float* xrow = x + (size_t)(rowBase + nrow) * DDIM;
    const int   roff  = (lane & 16) ? 8 : 0;       // D rows this lane holds

    float    psum[8];
    #pragma unroll
    for (int v = 0; v < 8; ++v) psum[v] = 0.f;
    unsigned pnz = 0u;

    for (int rr = 0; rr < 2; ++rr) {
        const int r      = wave * 2 + rr;
        const int kbase0 = r * BSZ;

        // Preload A fragments (scaled x -> f16), reused across 16 n-tiles.
        v16h afrag[8];
        #pragma unroll
        for (int kk = 0; kk < 8; ++kk) {
            const int c0 = kbase0 + kk * 32 + loffA;
            float4 p0 = *(const float4*)(xrow + c0);
            float4 p1 = *(const float4*)(xrow + c0 + 4);
            float4 p2 = *(const float4*)(xrow + c0 + 16);
            float4 p3 = *(const float4*)(xrow + c0 + 20);
            v16h a;
            a[0]  = (_Float16)(sA * p0.x); a[1]  = (_Float16)(sA * p0.y);
            a[2]  = (_Float16)(sA * p0.z); a[3]  = (_Float16)(sA * p0.w);
            a[4]  = (_Float16)(sA * p1.x); a[5]  = (_Float16)(sA * p1.y);
            a[6]  = (_Float16)(sA * p1.z); a[7]  = (_Float16)(sA * p1.w);
            a[8]  = (_Float16)(sA * p2.x); a[9]  = (_Float16)(sA * p2.y);
            a[10] = (_Float16)(sA * p2.z); a[11] = (_Float16)(sA * p2.w);
            a[12] = (_Float16)(sA * p3.x); a[13] = (_Float16)(sA * p3.y);
            a[14] = (_Float16)(sA * p3.z); a[15] = (_Float16)(sA * p3.w);
            afrag[kk] = a;
        }

        const _Float16* wblk = Wh + (size_t)r * BSZ * BSZ;   // [j][k] row-major
        for (int nt = 0; nt < 16; ++nt) {
            const _Float16* wrow = wblk + (size_t)(nt * 16 + nrow) * BSZ;
            v8f acc = {};
            #pragma unroll
            for (int kk = 0; kk < 8; ++kk) {
                const int kb = kk * 32 + loffB;
                union { v16h v; v8h h[2]; } b;
                b.h[0] = *(const v8h*)(wrow + kb);
                b.h[1] = *(const v8h*)(wrow + kb + 16);
                acc = __builtin_amdgcn_wmma_f32_16x16x32_f16(
                        false, afrag[kk], false, b.v, (short)0, acc, false, false);
            }
            const int col = r * BSZ + nt * 16 + nrow;
            #pragma unroll
            for (int v = 0; v < 8; ++v) {
                float d = acc[v];
                mx[(v + roff) * DDIM + col] = d;
                psum[v] += d * d;
                pnz |= (d != 0.f) ? (1u << v) : 0u;
            }
        }
    }

    // ---- reduce ||mx||^2 per row across lanes/waves ----
    #pragma unroll
    for (int v = 0; v < 8; ++v) {
        psum[v] += __shfl_xor(psum[v], 1, 32);
        psum[v] += __shfl_xor(psum[v], 2, 32);
        psum[v] += __shfl_xor(psum[v], 4, 32);
        psum[v] += __shfl_xor(psum[v], 8, 32);
    }
    pnz |= __shfl_xor(pnz, 1, 32);
    pnz |= __shfl_xor(pnz, 2, 32);
    pnz |= __shfl_xor(pnz, 4, 32);
    pnz |= __shfl_xor(pnz, 8, 32);
    if ((lane & 15) == 0) {
        #pragma unroll
        for (int v = 0; v < 8; ++v) atomicAdd(&msum[v + roff], psum[v]);
        atomicOr(&mnz[0], pnz << roff);
    }
    __syncthreads();

    // ---- per-row epilogue scalar: tanh/project/logmap0 collapsed ----
    if (tid < 16) {
        float mxn   = sqrtf(msum[tid]);
        float scale = 0.f;
        if (((mnz[0] >> tid) & 1u) && mxn > 0.f) {
            float t   = tanhf(fminf(alpha_[tid] * mxn, 15.f));
            float nrm = t / SQRTC;
            float p   = fminf(1.f, 9.99f / fmaxf(nrm, 1e-5f)); // (1-1e-3)/sqrt(c)
            float yn  = fmaxf(p * nrm, 1e-5f);
            float z   = fminf(SQRTC * yn, 1.f - 1e-5f);
            float at  = 0.5f * (logf(1.f + z) - logf(1.f - z));
            scale = at / (SQRTC * mxn);
        }
        mscale[tid] = scale;
    }
    __syncthreads();

    // ---- scaled store, coalesced b128 ----
    {
        float* orow = out + (size_t)rowBase * DDIM;
        #pragma unroll 2
        for (int j = 0; j < 64; ++j) {
            int idx = (j * 256 + tid) * 4;
            int m   = idx >> 12;
            float4 v = *(const float4*)(mx + idx);
            float sc = mscale[m];
            v.x *= sc; v.y *= sc; v.z *= sc; v.w *= sc;
            *(float4*)(orow + idx) = v;
        }
    }
}

extern "C" void kernel_launch(void* const* d_in, const int* in_sizes, int n_in,
                              void* d_out, int out_size, void* d_ws, size_t ws_size,
                              hipStream_t stream) {
    const float* x = (const float*)d_in[0];
    const float* W = (const float*)d_in[1];
    float* out     = (float*)d_out;
    _Float16* Wh   = (_Float16*)d_ws;   // 2 MB f16 weights

    const int wn4 = (R_BLK * BSZ * BSZ) / 4;
    wconv_kernel<<<(wn4 + 255) / 256, 256, 0, stream>>>(W, Wh, wn4);

    const int rows = in_sizes[0] / DDIM;        // 8192
    const int grid = rows / MTILE;              // 512
    const size_t shmem = (size_t)MTILE * DDIM * 4 + 6 * 16 * 4 + 16;

    static bool attr_done = false;
    (void)attr_done; // attribute raise is idempotent & host-side (graph-safe)
    hipFuncSetAttribute((const void*)hyp_blockdiag_kernel,
                        hipFuncAttributeMaxDynamicSharedMemorySize, (int)shmem);

    hyp_blockdiag_kernel<<<grid, 256, shmem, stream>>>(x, Wh, out);
}
